// BiLSTMTagger_76020921139770
// MI455X (gfx1250) — matrix operand order repaired and checked by reading, not compiled, not run on hardware
//
#include <hip/hip_runtime.h>
#include <hip/hip_bf16.h>
#include <math.h>

// ---------------- problem constants ----------------
#define VOCABN 50000
#define TAGSN  50
#define EDIM   128
#define HDIM   256
#define BB     64
#define SSEQ   1024
#define G4H    1024    // 4*H
#define KCAT   384     // E + H
#define NTAGP  64      // TAGS padded to 64

typedef __bf16 bf16_t;
typedef __attribute__((ext_vector_type(16))) __bf16 v16bf;
typedef __attribute__((ext_vector_type(8)))  __bf16 v8bf;
typedef __attribute__((ext_vector_type(8)))  float  v8f;

__device__ __forceinline__ v16bf cat88(v8bf lo, v8bf hi) {
  return __builtin_shufflevector(lo, hi, 0,1,2,3,4,5,6,7,8,9,10,11,12,13,14,15);
}
__device__ __forceinline__ v8f splat8(float x) {
  v8f v = {x,x,x,x,x,x,x,x};
  return v;
}
__device__ __forceinline__ float sigmf(float x) {
  return 1.0f / (1.0f + __expf(-x));
}
// Opaque wave-uniform zero: defeats invariant-load hoisting (keeps weight
// streams in the loop hitting L1/L2 instead of being spilled to scratch).
__device__ __forceinline__ int opaque_zero() {
  int zr;
  asm volatile("s_mov_b32 %0, 0" : "=s"(zr));
  return zr;
}

// ---------------- prep kernels ----------------

// e_bf[(s*B + b)*E + k] = bf16(emb[x[b][s]][k])
__global__ void k_embed(const int* __restrict__ x, const float* __restrict__ emb,
                        unsigned short* __restrict__ e_u16) {
  bf16_t* e_bf = reinterpret_cast<bf16_t*>(e_u16);
  int n = blockIdx.x * blockDim.x + threadIdx.x;        // over B*S*E = 8388608
  if (n >= BB * SSEQ * EDIM) return;
  int k = n & (EDIM - 1);
  int b = (n >> 7) & (BB - 1);
  int s = n >> 13;
  int tok = x[b * SSEQ + s];
  e_bf[n] = (bf16_t)emb[(size_t)tok * EDIM + k];
}

// Wcat[dir][row 0..1023][col 0..383]: cols 0..127 = Wx[row][col], 128..383 = Wh[row][col-128]
__global__ void k_wcat(const float* __restrict__ Wx_f, const float* __restrict__ Wh_f,
                       const float* __restrict__ Wx_b, const float* __restrict__ Wh_b,
                       unsigned short* __restrict__ w_u16) {
  bf16_t* Wcat = reinterpret_cast<bf16_t*>(w_u16);
  int n = blockIdx.x * blockDim.x + threadIdx.x;        // over 2*1024*384 = 786432
  if (n >= 2 * G4H * KCAT) return;
  int dir = n / (G4H * KCAT);
  int rem = n - dir * (G4H * KCAT);
  int row = rem / KCAT;
  int col = rem - row * KCAT;
  const float* Wx = dir ? Wx_b : Wx_f;
  const float* Wh = dir ? Wh_b : Wh_f;
  float v = (col < EDIM) ? Wx[row * EDIM + col] : Wh[row * HDIM + (col - EDIM)];
  Wcat[n] = (bf16_t)v;
}

// WfcP[tag 0..63][k 0..511], zero-padded past tag 49
__global__ void k_wfc(const float* __restrict__ Wfc, unsigned short* __restrict__ w_u16) {
  bf16_t* W = reinterpret_cast<bf16_t*>(w_u16);
  int n = blockIdx.x * blockDim.x + threadIdx.x;        // over 64*512 = 32768
  if (n >= NTAGP * 2 * HDIM) return;
  int tag = n >> 9;
  int k = n & 511;
  float v = (tag < TAGSN) ? Wfc[tag * 2 * HDIM + k] : 0.0f;
  W[n] = (bf16_t)v;
}

// biasc[dir][g] = bx[g] + bh[g]
__global__ void k_bias(const float* __restrict__ bx_f, const float* __restrict__ bh_f,
                       const float* __restrict__ bx_b, const float* __restrict__ bh_b,
                       float* __restrict__ biasc) {
  int n = blockIdx.x * blockDim.x + threadIdx.x;        // over 2048
  if (n >= 2 * G4H) return;
  int dir = n >> 10;
  int g = n & 1023;
  biasc[n] = dir ? (bx_b[g] + bh_b[g]) : (bx_f[g] + bh_f[g]);
}

// ---------------- fused bidirectional LSTM scan ----------------
// grid: 8 blocks = {dir 0..1} x {mgroup 0..3}; block = 256 threads = 8 waves.
// Per step: gates[16,1024] = [e_t(16,128) | h(16,256)] x Wcat^T  (12 WMMA k-steps);
// wave w owns N-tiles {w + 8j, j=0..7} so its i/f/g/o gate columns align; cell
// state f32 in registers, h ping-pongs through LDS in bf16; weights streamed
// from L1/L2 every step (full residency is >6K VGPRs/WGP -> impossible).
__global__ __launch_bounds__(256) void k_scan(const unsigned short* __restrict__ e_u16,
                                              const unsigned short* __restrict__ w_u16,
                                              const float* __restrict__ biasc,
                                              unsigned short* __restrict__ out_u16) {
  const bf16_t* e_bf = reinterpret_cast<const bf16_t*>(e_u16);
  const bf16_t* Wcat = reinterpret_cast<const bf16_t*>(w_u16);
  bf16_t* out_bf = reinterpret_cast<bf16_t*>(out_u16);

  const int dir  = blockIdx.x >> 2;
  const int mg   = blockIdx.x & 3;
  const int lane = threadIdx.x & 31;
  const int wv   = threadIdx.x >> 5;      // 0..7
  const int ln15 = lane & 15;
  const int khal = lane >> 4;             // 0/1

  __shared__ bf16_t hbuf[2][16 * HDIM];   // ping-pong h, 16 KB
  for (int i = threadIdx.x; i < 2 * 16 * HDIM; i += 256)
    (&hbuf[0][0])[i] = (bf16_t)0.0f;
  __syncthreads();

  const bf16_t* Wd = Wcat + (size_t)dir * G4H * KCAT;

  // per-lane biases for the 8 owned N-tiles (column-only, splat across M)
  float bj[8];
#pragma unroll
  for (int j = 0; j < 8; ++j)
    bj[j] = biasc[dir * G4H + 16 * (wv + 8 * j) + ln15];

  v8f c0 = splat8(0.0f), c1 = splat8(0.0f);   // cell state, f32 in registers

  for (int stp = 0; stp < SSEQ; ++stp) {
    const int t = dir ? (SSEQ - 1 - stp) : stp;

    // opaque zero per iteration: weight loads must stay inside the loop
    const bf16_t* Wdt = Wd + opaque_zero();

    v8f acc[8];
#pragma unroll
    for (int j = 0; j < 8; ++j) acc[j] = splat8(bj[j]);

    const bf16_t* erow = e_bf + ((size_t)t * BB + mg * 16 + ln15) * EDIM;
    const bf16_t* hrow = &hbuf[stp & 1][ln15 * HDIM];

    // prefetch next step's embedding row (global_prefetch_b8)
    {
      int tn = dir ? (t > 0 ? t - 1 : 0) : (t < SSEQ - 1 ? t + 1 : SSEQ - 1);
      const bf16_t* ep = e_bf + ((size_t)tn * BB + mg * 16 + ln15) * EDIM;
      __builtin_prefetch(ep, 0, 3);
    }

#pragma unroll
    for (int ks = 0; ks < 12; ++ks) {
      // A fragment: row M = ln15; K = k0 + khalf*8 + {0..7}  and  +16 + khalf*8 + {0..7}
      v16bf afrag;
      if (ks < 4) {
        const int ck = ks * 32 + khal * 8;
        v8bf lo = *(const v8bf*)(erow + ck);
        v8bf hi = *(const v8bf*)(erow + ck + 16);
        afrag = cat88(lo, hi);
      } else {
        const int ck = ks * 32 - EDIM + khal * 8;
        v8bf lo = *(const v8bf*)(hrow + ck);
        v8bf hi = *(const v8bf*)(hrow + ck + 16);
        afrag = cat88(lo, hi);
      }
      // batch the 8 B-fragment loads (two b128 each: K = khalf*8 + {0..7}, +16),
      // then issue 8 independent WMMAs (acc chains only across k-steps)
      v16bf bfrag[8];
#pragma unroll
      for (int j = 0; j < 8; ++j) {
        const bf16_t* brow =
            Wdt + (size_t)(16 * (wv + 8 * j) + ln15) * KCAT + ks * 32 + khal * 8;
        bfrag[j] = cat88(*(const v8bf*)brow, *(const v8bf*)(brow + 16));
      }
#pragma unroll
      for (int j = 0; j < 8; ++j) {
        acc[j] = __builtin_amdgcn_wmma_f32_16x16x32_bf16(
            false, afrag, false, bfrag[j], (short)0, acc[j], false, false);
      }
    }

    // LSTM cell update: tiles j = {0,1}=i, {2,3}=f, {4,5}=g, {6,7}=o
    bf16_t* hw = &hbuf[(stp + 1) & 1][0];
#pragma unroll
    for (int p = 0; p < 2; ++p) {
      v8f& cs = p ? c1 : c0;
      const int hcol = 16 * wv + 128 * p + ln15;
#pragma unroll
      for (int r = 0; r < 8; ++r) {
        float iv = sigmf(acc[0 + p][r]);
        float fv = sigmf(acc[2 + p][r]);
        float gv = tanhf(acc[4 + p][r]);
        float ov = sigmf(acc[6 + p][r]);
        float cc = fv * cs[r] + iv * gv;
        cs[r] = cc;
        float hv = ov * tanhf(cc);
        const int mrow = r + 8 * khal;       // C-frag: VGPR r -> M = r + 8*khalf
        bf16_t hb = (bf16_t)hv;
        hw[mrow * HDIM + hcol] = hb;
        const int b = mg * 16 + mrow;
        out_bf[((size_t)b * SSEQ + t) * (2 * HDIM) + dir * HDIM + hcol] = hb;
      }
    }
    __syncthreads();
  }
}

// ---------------- FC head: [65536,512] x [512,64] via WMMA ----------------
// block = 256 thr = 8 waves; wave w: mtile = bx*2 + (w>>2), ntile = w&3; K = 16 steps.
__global__ __launch_bounds__(256) void k_fc(const unsigned short* __restrict__ out_u16,
                                            const unsigned short* __restrict__ wfc_u16,
                                            const float* __restrict__ bfc,
                                            float* __restrict__ logits) {
  const bf16_t* out_bf = reinterpret_cast<const bf16_t*>(out_u16);
  const bf16_t* WfcP = reinterpret_cast<const bf16_t*>(wfc_u16);

  const int lane = threadIdx.x & 31;
  const int wv   = threadIdx.x >> 5;
  const int ln15 = lane & 15;
  const int khal = lane >> 4;
  const int mtile = blockIdx.x * 2 + (wv >> 2);
  const int nt    = wv & 3;

  const bf16_t* arow = out_bf + ((size_t)mtile * 16 + ln15) * (2 * HDIM);
  v8f acc = splat8(0.0f);
#pragma unroll
  for (int ks = 0; ks < 16; ++ks) {
    const int ck = ks * 32 + khal * 8;
    v8bf lo = *(const v8bf*)(arow + ck);
    v8bf hi = *(const v8bf*)(arow + ck + 16);
    v16bf afrag = cat88(lo, hi);
    const bf16_t* brow = WfcP + (size_t)(nt * 16 + ln15) * (2 * HDIM) + ks * 32 + khal * 8;
    v16bf bfrag = cat88(*(const v8bf*)brow, *(const v8bf*)(brow + 16));
    acc = __builtin_amdgcn_wmma_f32_16x16x32_bf16(
        false, afrag, false, bfrag, (short)0, acc, false, false);
  }

  const int tag = nt * 16 + ln15;
  if (tag < TAGSN) {
    const float bb = bfc[tag];
#pragma unroll
    for (int r = 0; r < 8; ++r) {
      const size_t R = (size_t)mtile * 16 + r + 8 * khal;   // R = b*S + s
      logits[R * TAGSN + tag] = acc[r] + bb;
    }
  }
}

// ---------------- host launcher ----------------
extern "C" void kernel_launch(void* const* d_in, const int* in_sizes, int n_in,
                              void* d_out, int out_size, void* d_ws, size_t ws_size,
                              hipStream_t stream) {
  const int*   x    = (const int*)d_in[0];
  // d_in[1] = lengths: unused by the reference computation
  const float* emb  = (const float*)d_in[2];
  const float* Wx_f = (const float*)d_in[3];
  const float* bx_f = (const float*)d_in[4];
  const float* Wh_f = (const float*)d_in[5];
  const float* bh_f = (const float*)d_in[6];
  const float* Wx_b = (const float*)d_in[7];
  const float* bx_b = (const float*)d_in[8];
  const float* Wh_b = (const float*)d_in[9];
  const float* bh_b = (const float*)d_in[10];
  const float* Wfc  = (const float*)d_in[11];
  const float* bfc  = (const float*)d_in[12];
  float* logits = (float*)d_out;

  char* ws = (char*)d_ws;
  // workspace layout (256B aligned): e_bf | Wcat | WfcP | biasc | out_bf  (~85.5 MB)
  unsigned short* e_bf  = (unsigned short*)(ws + 0);          // 16,777,216 B
  unsigned short* Wcat  = (unsigned short*)(ws + 16777216);   //  1,572,864 B
  unsigned short* WfcP  = (unsigned short*)(ws + 18350080);   //     65,536 B
  float*          biasc = (float*)        (ws + 18415616);    //      8,192 B
  unsigned short* out_b = (unsigned short*)(ws + 18423808);   // 67,108,864 B

  // 1) embedding gather -> bf16, time-major rows (s*B + b)
  k_embed<<<(BB * SSEQ * EDIM + 255) / 256, 256, 0, stream>>>(x, emb, e_bf);
  // 2) concatenated recurrent weights [Wx | Wh] -> bf16
  k_wcat<<<(2 * G4H * KCAT + 255) / 256, 256, 0, stream>>>(Wx_f, Wh_f, Wx_b, Wh_b, Wcat);
  // 3) padded FC weights -> bf16
  k_wfc<<<(NTAGP * 2 * HDIM + 255) / 256, 256, 0, stream>>>(Wfc, WfcP);
  // 4) fused biases bx+bh
  k_bias<<<(2 * G4H + 255) / 256, 256, 0, stream>>>(bx_f, bh_f, bx_b, bh_b, biasc);
  // 5) bidirectional LSTM scan (8 persistent workgroups, WMMA per step)
  k_scan<<<8, 256, 0, stream>>>(e_bf, Wcat, biasc, out_b);
  // 6) FC head (WMMA), M-tiles = 4096, 2 per block
  k_fc<<<(BB * SSEQ / 16) / 2, 256, 0, stream>>>(out_b, WfcP, bfc, logits);
}
